// SpectralConv_14688788152384
// MI455X (gfx1250) — matrix operand order repaired and checked by reading, not compiled, not run on hardware
//
#include <hip/hip_runtime.h>
#include <hip/hip_bf16.h>

// ---------------------------------------------------------------------------
// SpectralConv (FNO) on MI455X / gfx1250, wave32 + WMMA bf16 (f32 accum).
//
//   x: (8,256,256,64) f32   w_real/w_imag: (16,16,64,64) f32   out: f32
//
// Truncated-DFT-as-GEMM pipeline on v_wmma_f32_16x16x32_bf16:
//   K0  build twiddle tables PRE-PACKED in WMMA A-fragment lane layout (bf16)
//   K1  DFT along H:  Xh[b,k1,(w,c)]  M=16 K=256 N=16384   (real->complex)
//   K2  DFT along W:  Xm[b,k1,k2,c]   M=16 K=256 N=64      (complex)
//   K3  channel mix:  Y[b,k1,k2,o]    M=16(pad b) K=64 N=64 (complex, w*0.02)
//   K4  fused inverse per (b, 16-row h tile):
//         stage A: Z = E(h, k1-re|im) . [Yr;Yi]  (single K=32 WMMA)
//                  -> LDS, pre-packed as bf16 B-fragments (64 KB)
//         stage B: out = C(w, k2-re|im) . Z      (irfft coeffs + 1/256^2 folded)
//
// Intermediates are stored bf16: identical rounding to converting at the next
// WMMA load, but half the traffic and no cvt on the consumer side.
// ---------------------------------------------------------------------------

typedef __bf16  v16bf __attribute__((ext_vector_type(16)));
typedef float   v8f   __attribute__((ext_vector_type(8)));

// ---- workspace layout (BYTE offsets) --------------------------------------
constexpr size_t OFF_PFC  = 0;                       // fwd cos  [8 steps][32][16] bf16
constexpr size_t OFF_PFS  = 8192;                    // fwd -sin
constexpr size_t OFF_PFSN = 16384;                   // fwd +sin
constexpr size_t OFF_PAZR = 24576;                   // inv-H re [16 tiles][32][16] bf16
constexpr size_t OFF_PAZI = 40960;                   // inv-H im
constexpr size_t OFF_PCF  = 57344;                   // irfft synthesis (scaled)
constexpr size_t OFF_XHRE = 73728;                   // [8][16][16384] bf16 = 4 MB
constexpr size_t OFF_XHIM = OFF_XHRE + 4194304;
constexpr size_t OFF_XMRE = OFF_XHIM + 4194304;      // [8][16][16][64] bf16
constexpr size_t OFF_XMIM = OFF_XMRE + 262144;
constexpr size_t OFF_YRE  = OFF_XMIM + 262144;
constexpr size_t OFF_YIM  = OFF_YRE  + 262144;
// total ~ 9.6 MB of d_ws

// ---- WMMA helpers (ISA 7.12.2 layouts, wave32) -----------------------------
__device__ __forceinline__ v8f wmma_bf16(v16bf a, v16bf b, v8f c) {
  return __builtin_amdgcn_wmma_f32_16x16x32_bf16(
      /*neg_a=*/false, a, /*neg_b=*/false, b,
      /*c_mod=*/(short)0, c, /*reuse_a=*/false, /*reuse_b=*/false);
}

// A fragment pre-packed by K0: one contiguous 32B load per lane.
__device__ __forceinline__ v16bf load_a_packed(const __bf16* __restrict__ t, int tile) {
  int l = threadIdx.x & 31;
  return *(const v16bf*)(t + (((size_t)tile * 32 + l) << 4));
}

// A matrix 16x32 from bf16 src, rows >= 8 zeroed (batch pad), optional negate.
__device__ __forceinline__ v16bf load_a_pad8_bf16(const __bf16* __restrict__ src,
                                                  int rs, bool neg) {
  int l = threadIdx.x & 31, m = l & 15, g = l >> 4;
  v16bf a;
#pragma unroll
  for (int i = 0; i < 16; ++i) {
    int k = i + 16 * g;
    float v = (m < 8) ? (float)src[m * rs + k] : 0.0f;
    a[i] = (__bf16)(neg ? -v : v);
  }
  return a;
}

// B matrix 32x16: lane n = l&15, half i -> K = i + 16g.
__device__ __forceinline__ v16bf load_b_f32(const float* __restrict__ src, int rs) {
  int l = threadIdx.x & 31, n = l & 15, g = l >> 4;
  v16bf b;
#pragma unroll
  for (int i = 0; i < 16; ++i) b[i] = (__bf16)src[(i + 16 * g) * rs + n];
  return b;
}

__device__ __forceinline__ v16bf load_b_bf16(const __bf16* __restrict__ src, int rs) {
  int l = threadIdx.x & 31, n = l & 15, g = l >> 4;
  v16bf b;
#pragma unroll
  for (int i = 0; i < 16; ++i) b[i] = src[(i + 16 * g) * rs + n];
  return b;
}

__device__ __forceinline__ v16bf load_b_scaled(const float* __restrict__ src,
                                               int rs, float scale) {
  int l = threadIdx.x & 31, n = l & 15, g = l >> 4;
  v16bf b;
#pragma unroll
  for (int i = 0; i < 16; ++i) b[i] = (__bf16)(src[(i + 16 * g) * rs + n] * scale);
  return b;
}

// B packing [Yre(16) ; Yim(16)] along K for the complex contraction (bf16 src).
__device__ __forceinline__ v16bf load_b_split_bf16(const __bf16* __restrict__ re,
                                                   const __bf16* __restrict__ im,
                                                   int rs, int colb) {
  int l = threadIdx.x & 31, n = l & 15, g = l >> 4;
  v16bf b;
#pragma unroll
  for (int i = 0; i < 16; ++i) {
    int k = i + 16 * g;
    b[i] = (k < 16) ? re[k * rs + colb + n] : im[(k - 16) * rs + colb + n];
  }
  return b;
}

// C/D 16x16 f32: element r -> row M = r + 8g, col N = l&15.
__device__ __forceinline__ void store_c(float* __restrict__ dst, int rs, v8f c) {
  int l = threadIdx.x & 31, n = l & 15, g = l >> 4;
#pragma unroll
  for (int r = 0; r < 8; ++r) dst[(r + 8 * g) * rs + n] = c[r];
}

__device__ __forceinline__ void store_c_bf16(__bf16* __restrict__ dst, int rs, v8f c) {
  int l = threadIdx.x & 31, n = l & 15, g = l >> 4;
#pragma unroll
  for (int r = 0; r < 8; ++r) dst[(r + 8 * g) * rs + n] = (__bf16)c[r];
}

__device__ __forceinline__ void store_c_rows8_bf16(__bf16* __restrict__ dst, int rs, v8f c) {
  int l = threadIdx.x & 31, n = l & 15, g = l >> 4;
  if (g == 0) {
#pragma unroll
    for (int r = 0; r < 8; ++r) dst[r * rs + n] = (__bf16)c[r];
  }
}

// ---- K0: twiddle tables, emitted directly in packed A-fragment layout ------
__global__ void __launch_bounds__(256) k_tables(char* __restrict__ wsb) {
  __bf16* PFC  = (__bf16*)(wsb + OFF_PFC);
  __bf16* PFS  = (__bf16*)(wsb + OFF_PFS);
  __bf16* PFSN = (__bf16*)(wsb + OFF_PFSN);
  __bf16* PAZR = (__bf16*)(wsb + OFF_PAZR);
  __bf16* PAZI = (__bf16*)(wsb + OFF_PAZI);
  __bf16* PCF  = (__bf16*)(wsb + OFF_PCF);
  const float STEP = 0.024543692603f;  // 2*pi/256
  int tid = blockIdx.x * blockDim.x + threadIdx.x;
  int nth = gridDim.x * blockDim.x;
  // forward: A rows m = mode (16), K = h/w sample; exp(-i*th) = cos - i*sin
  for (int idx = tid; idx < 4096; idx += nth) {       // [8 steps][32 lanes][16]
    int step = idx >> 9, lane = (idx >> 4) & 31, i = idx & 15;
    int m = lane & 15, g = lane >> 4;
    int k = step * 32 + i + 16 * g;
    float th = (float)((m * k) & 255) * STEP;
    float s = __sinf(th), c = __cosf(th);
    PFC[idx]  = (__bf16)c;
    PFS[idx]  = (__bf16)(-s);
    PFSN[idx] = (__bf16)s;
  }
  // inverse: A rows = h (AZR/AZI) or w (CF); K=32 packs j<16 (re), j>=16 (im)
  for (int idx = tid; idx < 8192; idx += nth) {       // [16 tiles][32 lanes][16]
    int tile = idx >> 9, lane = (idx >> 4) & 31, i = idx & 15;
    int m = lane & 15, g = lane >> 4;
    int row = tile * 16 + m;
    int j = i + 16 * g;
    int k = j & 15;
    float th = (float)((k * row) & 255) * STEP;
    float s = __sinf(th), c = __cosf(th);
    PAZR[idx] = (__bf16)((j < 16) ? c : -s);          // Zre = cos*Yr - sin*Yi
    PAZI[idx] = (__bf16)((j < 16) ? s :  c);          // Zim = sin*Yr + cos*Yi
    float amp = ((k == 0) ? 1.0f : 2.0f) * (1.0f / 65536.0f);  // irfft + 1/N^2
    PCF[idx] = (__bf16)((j < 16) ? amp * c : ((k == 0) ? 0.0f : -amp * s));
  }
}

// ---- K1: DFT along H (real -> complex), M=16 K=256 N=16384 per batch -------
__global__ void __launch_bounds__(256) k_dft_h(const float* __restrict__ x,
                                               char* __restrict__ wsb) {
  const __bf16* PFC = (const __bf16*)(wsb + OFF_PFC);
  const __bf16* PFS = (const __bf16*)(wsb + OFF_PFS);
  __bf16* XhRe = (__bf16*)(wsb + OFF_XHRE);
  __bf16* XhIm = (__bf16*)(wsb + OFF_XHIM);
  int tile = blockIdx.x * (blockDim.x >> 5) + (threadIdx.x >> 5);  // 8192 tiles
  int b    = tile >> 10;
  int colb = (tile & 1023) * 16;                 // column = w*64 + c
  const float* xb = x + (size_t)b * 256 * 16384 + colb;
  v8f accR = {}, accI = {};
#pragma unroll 2
  for (int s = 0; s < 8; ++s) {                  // K-steps of 32 over h
    v16bf bf = load_b_f32(xb + (size_t)s * 32 * 16384, 16384);
    v16bf ac = load_a_packed(PFC, s);
    v16bf as = load_a_packed(PFS, s);
    accR = wmma_bf16(ac, bf, accR);
    accI = wmma_bf16(as, bf, accI);
  }
  store_c_bf16(XhRe + (size_t)b * 16 * 16384 + colb, 16384, accR);
  store_c_bf16(XhIm + (size_t)b * 16 * 16384 + colb, 16384, accI);
}

// ---- K2: DFT along W (complex), per (b,k1): M=16 K=256 N=64 ----------------
__global__ void __launch_bounds__(256) k_dft_w(char* __restrict__ wsb) {
  const __bf16* PFC  = (const __bf16*)(wsb + OFF_PFC);
  const __bf16* PFS  = (const __bf16*)(wsb + OFF_PFS);
  const __bf16* PFSN = (const __bf16*)(wsb + OFF_PFSN);
  const __bf16* XhRe = (const __bf16*)(wsb + OFF_XHRE);
  const __bf16* XhIm = (const __bf16*)(wsb + OFF_XHIM);
  __bf16* XmRe = (__bf16*)(wsb + OFF_XMRE);
  __bf16* XmIm = (__bf16*)(wsb + OFF_XMIM);
  int tile = blockIdx.x * (blockDim.x >> 5) + (threadIdx.x >> 5);  // 512 tiles
  int b  = tile >> 6;
  int k1 = (tile >> 2) & 15;
  int nt = tile & 3;
  const __bf16* br = XhRe + (size_t)(b * 16 + k1) * 16384 + nt * 16;
  const __bf16* bi = XhIm + (size_t)(b * 16 + k1) * 16384 + nt * 16;
  v8f accR = {}, accI = {};
  for (int s = 0; s < 8; ++s) {                  // K-steps of 32 over w
    v16bf ac  = load_a_packed(PFC,  s);
    v16bf as  = load_a_packed(PFS,  s);
    v16bf asn = load_a_packed(PFSN, s);
    v16bf vr = load_b_bf16(br + s * 32 * 64, 64);
    v16bf vi = load_b_bf16(bi + s * 32 * 64, 64);
    accR = wmma_bf16(ac,  vr, accR);   // re += cos*vr
    accR = wmma_bf16(asn, vi, accR);   // re += sin*vi
    accI = wmma_bf16(ac,  vi, accI);   // im += cos*vi
    accI = wmma_bf16(as,  vr, accI);   // im += -sin*vr
  }
  store_c_bf16(XmRe + (size_t)(b * 16 + k1) * 1024 + nt * 16, 64, accR);
  store_c_bf16(XmIm + (size_t)(b * 16 + k1) * 1024 + nt * 16, 64, accI);
}

// ---- K3: complex channel mix per (k1,k2): (8->16)x64x64, weights * 0.02 ----
__global__ void __launch_bounds__(128) k_mix(const float* __restrict__ wre,
                                             const float* __restrict__ wim,
                                             char* __restrict__ wsb) {
  const float W_SCALE = 0.02f;
  const __bf16* XmRe = (const __bf16*)(wsb + OFF_XMRE);
  const __bf16* XmIm = (const __bf16*)(wsb + OFF_XMIM);
  __bf16* YRe = (__bf16*)(wsb + OFF_YRE);
  __bf16* YIm = (__bf16*)(wsb + OFF_YIM);
  int k1 = blockIdx.x >> 4, k2 = blockIdx.x & 15;
  int nt = threadIdx.x >> 5;                       // 4 waves = 4 o-tiles
  const __bf16* axr = XmRe + (size_t)k1 * 1024 + k2 * 64;  // b stride 16384
  const __bf16* axi = XmIm + (size_t)k1 * 1024 + k2 * 64;
  const float* bwr = wre + (size_t)(k1 * 16 + k2) * 4096 + nt * 16;
  const float* bwi = wim + (size_t)(k1 * 16 + k2) * 4096 + nt * 16;
  v8f accR = {}, accI = {};
#pragma unroll
  for (int k0 = 0; k0 < 64; k0 += 32) {
    v16bf ar  = load_a_pad8_bf16(axr + k0, 16384, false);
    v16bf ai  = load_a_pad8_bf16(axi + k0, 16384, false);
    v16bf ain = load_a_pad8_bf16(axi + k0, 16384, true);
    v16bf br  = load_b_scaled(bwr + k0 * 64, 64, W_SCALE);
    v16bf bi  = load_b_scaled(bwi + k0 * 64, 64, W_SCALE);
    accR = wmma_bf16(ar,  br, accR);   // re += Xr*Wr
    accR = wmma_bf16(ain, bi, accR);   // re -= Xi*Wi
    accI = wmma_bf16(ar,  bi, accI);   // im += Xr*Wi
    accI = wmma_bf16(ai,  br, accI);   // im += Xi*Wr
  }
  store_c_rows8_bf16(YRe + (size_t)k1 * 1024 + k2 * 64 + nt * 16, 16384, accR);
  store_c_rows8_bf16(YIm + (size_t)k1 * 1024 + k2 * 64 + nt * 16, 16384, accI);
}

// ---- K4: fused inverse; block = (b, 16-row h tile), 8 waves, 64 KB LDS -----
// LDS holds Z pre-packed as bf16 B-fragments: [h][o-tile][lane][16].
__global__ void __launch_bounds__(256) k_inverse(float* __restrict__ out,
                                                 const char* __restrict__ wsb) {
  extern __shared__ __bf16 ZP[];                     // 16*4*32*16 bf16 = 64 KB
  const __bf16* PAZR = (const __bf16*)(wsb + OFF_PAZR);
  const __bf16* PAZI = (const __bf16*)(wsb + OFF_PAZI);
  const __bf16* PCF  = (const __bf16*)(wsb + OFF_PCF);
  const __bf16* YRe  = (const __bf16*)(wsb + OFF_YRE);
  const __bf16* YIm  = (const __bf16*)(wsb + OFF_YIM);
  int b = blockIdx.x >> 4;
  int ht = blockIdx.x & 15;
  int wave = threadIdx.x >> 5;
  int l = threadIdx.x & 31, n = l & 15, g = l >> 4;
  // stage A: Z[h,(k2,o)] over K=32 = [16 k1 re | 16 k1 im]
  v16bf azr = load_a_packed(PAZR, ht);               // depends only on h tile
  v16bf azi = load_a_packed(PAZI, ht);
  const __bf16* yr = YRe + (size_t)b * 16384;        // [k1][k2*64+o], stride 1024
  const __bf16* yi = YIm + (size_t)b * 16384;
  for (int nt = wave; nt < 64; nt += 8) {            // 64 tiles over (k2,o)
    int colb = nt * 16;
    v16bf bf = load_b_split_bf16(yr, yi, 1024, colb);
    v8f zr = {}, zi = {};
    zr = wmma_bf16(azr, bf, zr);
    zi = wmma_bf16(azi, bf, zi);
    int col = colb + n, k2 = col >> 6, o = col & 63;
    int ot = o >> 4, nr = o & 15;
#pragma unroll
    for (int r = 0; r < 8; ++r) {                    // scatter into B-frag layout
      int h = r + 8 * g;
      int base = (h * 4 + ot) * 32;
      ZP[(size_t)(base + nr)      * 16 + k2] = (__bf16)zr[r];   // j = k2 (re)
      ZP[(size_t)(base + nr + 16) * 16 + k2] = (__bf16)zi[r];   // j = k2+16 (im)
    }
  }
  __syncthreads();
  // stage B: out[h,w,o] = CF(w,j) . Z(h,j,o); B-frag = contiguous 32B LDS load
  float* ob = out + ((size_t)b * 256 + ht * 16) * 16384;
  for (int p = wave; p < 64; p += 8) {               // (h, o-tile) pairs
    int h = p >> 2, ot = p & 3;
    v16bf bf = *(const v16bf*)(ZP + (size_t)((h * 4 + ot) * 32 + l) * 16);
#pragma unroll
    for (int mt = 0; mt < 16; ++mt) {                // 16 w-tiles of 16
      v16bf af = load_a_packed(PCF, mt);
      v8f acc = {};
      acc = wmma_bf16(af, bf, acc);
      store_c(ob + ((size_t)h * 256 + mt * 16) * 64 + ot * 16, 64, acc);
    }
  }
}

// ---------------------------------------------------------------------------
extern "C" void kernel_launch(void* const* d_in, const int* in_sizes, int n_in,
                              void* d_out, int out_size, void* d_ws, size_t ws_size,
                              hipStream_t stream) {
  const float* x  = (const float*)d_in[0];
  const float* wr = (const float*)d_in[1];
  const float* wi = (const float*)d_in[2];
  float* out = (float*)d_out;
  char*  ws  = (char*)d_ws;   // needs ~9.6 MB
  (void)in_sizes; (void)n_in; (void)out_size; (void)ws_size;

  k_tables <<<16,   256, 0,     stream>>>(ws);
  k_dft_h  <<<1024, 256, 0,     stream>>>(x, ws);
  k_dft_w  <<<64,   256, 0,     stream>>>(ws);
  k_mix    <<<256,  128, 0,     stream>>>(wr, wi, ws);
  k_inverse<<<128,  256, 65536, stream>>>(out, ws);
}
